// SentimentClassifier_21930103014176
// MI455X (gfx1250) — compile-verified, matrix operand order checked by reading, not run on hardware
//
#include <hip/hip_runtime.h>

typedef __bf16 bf16;
typedef __attribute__((ext_vector_type(16))) __bf16 v16bf;
typedef __attribute__((ext_vector_type(8)))  float  v8f;

#define NEGV (-1.0e9f)

// Async global->LDS copy (CDNA5). Inline asm avoids the builtin's HIP
// language-address-space pointer typing (param1 is 'int4 __device__*').
// ISA 15.18.3 op 98: global_load_async_to_lds_b128 vdst(=LDS addr), vaddr, off
#if defined(__HIP_DEVICE_COMPILE__) && defined(__gfx1250__) &&                \
    __has_builtin(__builtin_amdgcn_s_wait_asynccnt)
#define USE_ASYNC_LDS 1
#else
#define USE_ASYNC_LDS 0
#endif

#if USE_ASYNC_LDS
__device__ inline void async_cp16(const bf16* gp, bf16* lp) {
  // Generic pointer to a __shared__ object: low 32 bits are the LDS offset
  // (flat->LDS aperture: the LDS address is the low 32 bits).
  unsigned loff = (unsigned)(size_t)lp;
  asm volatile("global_load_async_to_lds_b128 %0, %1, off"
               :
               : "v"(loff), "v"(gp)
               : "memory");
}
__device__ inline void async_wait0() { __builtin_amdgcn_s_wait_asynccnt(0); }
#else
__device__ inline void async_cp16(const bf16* gp, bf16* lp) {
  *(uint4*)lp = *(const uint4*)gp;
}
__device__ inline void async_wait0() {}
#endif

// ---------------------------------------------------------------- WMMA helpers
__device__ inline v8f wmma_bf16(v16bf a, v16bf b, v8f c) {
  return __builtin_amdgcn_wmma_f32_16x16x32_bf16(false, a, false, b, (short)0, c,
                                                 false, false);
}

// Fragment load from LDS (row-major, stride ld, 16B-aligned rows).
// Per ISA 7.12.2 the per-lane 16 bf16 are two contiguous 8-element runs:
// [8*hsel .. 8*hsel+7] and [16+8*hsel .. 16+8*hsel+7]  ->  2x ds_load_b128.
__device__ inline v16bf frag_ld(const bf16* sm, int ld, int row0, int lane) {
  int m = row0 + (lane & 15);
  int hsel = (lane >> 4) & 1;
  const bf16* p = sm + m * ld + 8 * hsel;
  union {
    uint4 u[2];
    v16bf v;
  } t;
  t.u[0] = *(const uint4*)(p);
  t.u[1] = *(const uint4*)(p + 16);
  return t.v;
}

__device__ inline float gelu_exact(float x) {
  return 0.5f * x * (1.0f + erff(x * 0.70710678118654752f));
}

// ---------------------------------------------------------------- fp32 -> bf16
__global__ __launch_bounds__(256) void f32_to_bf16_kernel(
    const float* __restrict__ in, bf16* __restrict__ out, size_t n) {
  size_t i = (size_t)blockIdx.x * blockDim.x + threadIdx.x;
  size_t stride = (size_t)gridDim.x * blockDim.x;
  for (; i < n; i += stride) out[i] = (bf16)in[i];
}

// ---------------------------------------------------------------- generic GEMM
// C[M,N] = act( (A[M,K](bf16) @ B[K,N](bf16) + bias[N]) * alpha )
// 256 threads = 8 wave32, tile 128(M) x 64(N), K-steps of 32.
// B is staged TRANSPOSED into LDS so both fragments are ds_load_b128 pairs.
__global__ __launch_bounds__(256) void gemm_bf16_kernel(
    const bf16* __restrict__ A, int lda, const bf16* __restrict__ Bm, int ldb,
    const float* __restrict__ bias, float* __restrict__ Cf,
    bf16* __restrict__ Cb, int ldc, int M, int N, int K, float alpha, int act) {
  __shared__ alignas(16) bf16 As[128][40];
  __shared__ alignas(16) bf16 BsT[64][40];
  int tid = threadIdx.x, lane = tid & 31, w = tid >> 5;
  int wm = w & 3, wn = w >> 2;
  int rowBase = blockIdx.y * 128, colBase = blockIdx.x * 64;
  bool interior = (rowBase + 128 <= M) && (colBase + 64 <= N) && ((K & 31) == 0);

  v8f acc[2][2] = {};
  for (int k0 = 0; k0 < K; k0 += 32) {
    if (interior) {
      // A tile: 128x32 bf16 = 512 x 16B; 2 async 16B copies per thread.
#pragma unroll
      for (int it = 0; it < 2; ++it) {
        int idx = tid + it * 256;
        int r = idx >> 2, q = idx & 3;
        async_cp16(A + (size_t)(rowBase + r) * lda + k0 + q * 8, &As[r][q * 8]);
      }
      // B tile: 32x64, vector-load 8 cols of one k-row, scatter transposed.
      {
        int kR = tid >> 3, q = tid & 7;
        union {
          uint4 u;
          bf16 h[8];
        } t;
        t.u = *(const uint4*)(Bm + (size_t)(k0 + kR) * ldb + colBase + q * 8);
#pragma unroll
        for (int e = 0; e < 8; ++e) BsT[q * 8 + e][kR] = t.h[e];
      }
      // prefetch next K tile of A (global_prefetch_b8)
      if (k0 + 32 < K)
        __builtin_prefetch(A + (size_t)(rowBase + (tid & 127)) * lda + k0 + 32,
                           0, 1);
      async_wait0();
    } else {
      // branchless clamped path for edge tiles (e.g. M=2 qg projection)
      for (int i = tid; i < 128 * 32; i += 256) {
        int r = i >> 5, c = i & 31;
        int gr = rowBase + r, gc = k0 + c;
        int grc = gr < M ? gr : M - 1;
        int gcc = gc < K ? gc : K - 1;
        bf16 vv = A[(size_t)grc * lda + gcc];
        As[r][c] = (gr < M && gc < K) ? vv : (bf16)0.0f;
      }
      for (int i = tid; i < 32 * 64; i += 256) {
        int r = i >> 6, c = i & 63;
        int gr = k0 + r, gc = colBase + c;
        int grc = gr < K ? gr : K - 1;
        int gcc = gc < N ? gc : N - 1;
        bf16 vv = Bm[(size_t)grc * ldb + gcc];
        BsT[c][r] = (gr < K && gc < N) ? vv : (bf16)0.0f;
      }
    }
    __syncthreads();
    v16bf af0 = frag_ld(&As[0][0], 40, wm * 32, lane);
    v16bf af1 = frag_ld(&As[0][0], 40, wm * 32 + 16, lane);
    v16bf bf0 = frag_ld(&BsT[0][0], 40, wn * 32, lane);
    v16bf bf1 = frag_ld(&BsT[0][0], 40, wn * 32 + 16, lane);
    acc[0][0] = wmma_bf16(af0, bf0, acc[0][0]);
    acc[0][1] = wmma_bf16(af0, bf1, acc[0][1]);
    acc[1][0] = wmma_bf16(af1, bf0, acc[1][0]);
    acc[1][1] = wmma_bf16(af1, bf1, acc[1][1]);
    __syncthreads();
  }
  int nl = lane & 15, hsel = lane >> 4;
#pragma unroll
  for (int i = 0; i < 2; ++i)
#pragma unroll
    for (int j = 0; j < 2; ++j)
#pragma unroll
      for (int v = 0; v < 8; ++v) {
        int m = rowBase + wm * 32 + 16 * i + v + 8 * hsel;
        int n = colBase + wn * 32 + 16 * j + nl;
        if (interior || (m < M && n < N)) {
          float val = acc[i][j][v];
          if (bias) val += bias[n];
          val *= alpha;
          if (act == 1) val = gelu_exact(val);
          if (Cf) Cf[(size_t)m * ldc + n] = val;
          if (Cb) Cb[(size_t)m * ldc + n] = (bf16)val;
        }
      }
}

// ---------------------------------------------------------------- embed + LN
__global__ __launch_bounds__(256) void embed_ln_kernel(
    const int* __restrict__ src, const float* __restrict__ etok,
    const float* __restrict__ epos, const float* __restrict__ g,
    const float* __restrict__ bt, float* __restrict__ hf,
    bf16* __restrict__ hb, int S_, int D_) {
  int row = blockIdx.x;
  int s = row % S_;
  int tid = threadIdx.x;
  int tok = src[row];
  __shared__ float red[256];
  float x[4];
  float sum = 0.f;
  for (int d = tid, i = 0; d < D_; d += 256, ++i) {
    float v = etok[(size_t)tok * D_ + d] + epos[(size_t)s * D_ + d];
    x[i] = v;
    sum += v;
  }
  red[tid] = sum;
  __syncthreads();
  if (tid == 0) { float t = 0; for (int i = 0; i < 256; ++i) t += red[i]; red[0] = t; }
  __syncthreads();
  float mean = red[0] / (float)D_;
  __syncthreads();
  float vs = 0.f;
  for (int d = tid, i = 0; d < D_; d += 256, ++i) { float dv = x[i] - mean; vs += dv * dv; }
  red[tid] = vs;
  __syncthreads();
  if (tid == 0) { float t = 0; for (int i = 0; i < 256; ++i) t += red[i]; red[0] = t; }
  __syncthreads();
  float rstd = rsqrtf(red[0] / (float)D_ + 1e-5f);
  for (int d = tid, i = 0; d < D_; d += 256, ++i) {
    float v = (x[i] - mean) * rstd * g[d] + bt[d];
    hf[(size_t)row * D_ + d] = v;
    hb[(size_t)row * D_ + d] = (bf16)v;
  }
}

// ---------------------------------------------------------------- residual+LN
__global__ __launch_bounds__(256) void add_ln_kernel(
    const float* __restrict__ hin, const float* __restrict__ t,
    const float* __restrict__ g, const float* __restrict__ bt,
    float* __restrict__ hout, bf16* __restrict__ hb, int D_) {
  int row = blockIdx.x;
  int tid = threadIdx.x;
  __shared__ float red[256];
  float x[4];
  float sum = 0.f;
  for (int d = tid, i = 0; d < D_; d += 256, ++i) {
    float v = hin[(size_t)row * D_ + d] + t[(size_t)row * D_ + d];
    x[i] = v;
    sum += v;
  }
  red[tid] = sum;
  __syncthreads();
  if (tid == 0) { float s = 0; for (int i = 0; i < 256; ++i) s += red[i]; red[0] = s; }
  __syncthreads();
  float mean = red[0] / (float)D_;
  __syncthreads();
  float vs = 0.f;
  for (int d = tid, i = 0; d < D_; d += 256, ++i) { float dv = x[i] - mean; vs += dv * dv; }
  red[tid] = vs;
  __syncthreads();
  if (tid == 0) { float s = 0; for (int i = 0; i < 256; ++i) s += red[i]; red[0] = s; }
  __syncthreads();
  float rstd = rsqrtf(red[0] / (float)D_ + 1e-5f);
  for (int d = tid, i = 0; d < D_; d += 256, ++i) {
    float v = (x[i] - mean) * rstd * g[d] + bt[d];
    hout[(size_t)row * D_ + d] = v;
    hb[(size_t)row * D_ + d] = (bf16)v;
  }
}

// ---------------------------------------------------------------- QK^T band
// per (b,h,chunk): scores(128 x 384) = Qc(128 x 64) @ Kband^T(64 x 384)
__global__ __launch_bounds__(256) void qk_scores_kernel(
    const bf16* __restrict__ qb, const bf16* __restrict__ kb,
    float* __restrict__ scores, int B_, int S_, int D_, int Hn) {
  const int Wn = 128, KW = 384;
  int nc = S_ / Wn;
  int nt = blockIdx.x;  // N tile of 64 within 384
  int bhc = blockIdx.y;
  int c = bhc % nc, bh = bhc / nc, hh = bh % Hn, b = bh / Hn;
  int tid = threadIdx.x, lane = tid & 31, w = tid >> 5;
  int wm = w & 3, wn = w >> 2;
  __shared__ alignas(16) bf16 As[128][40];
  __shared__ alignas(16) bf16 BsT[64][40];
  v8f acc[2][2] = {};
  for (int k0 = 0; k0 < 64; k0 += 32) {
    // A: Q chunk rows, contiguous along dh
#pragma unroll
    for (int it = 0; it < 2; ++it) {
      int idx = tid + it * 256;
      int r = idx >> 2, q = idx & 3;
      int s = c * Wn + r;
      async_cp16(qb + (size_t)(b * S_ + s) * D_ + hh * 64 + k0 + q * 8,
                 &As[r][q * 8]);
    }
    // B^T: BsT[key][kdim] = K[b, gk(key), hh, k0+kdim]; dh contiguous -> vector
    {
      int n = tid >> 2, q = tid & 3;  // 64 keys x 4 chunks of 8
      int gk = (c - 1) * Wn + nt * 64 + n;
      int gkc = gk < 0 ? 0 : (gk >= S_ ? S_ - 1 : gk);
      uint4 t =
          *(const uint4*)(kb + (size_t)(b * S_ + gkc) * D_ + hh * 64 + k0 + q * 8);
      uint4 z = {0u, 0u, 0u, 0u};
      bool ok = (gk >= 0) && (gk < S_);
      *(uint4*)&BsT[n][q * 8] = ok ? t : z;
    }
    async_wait0();
    __syncthreads();
    v16bf af0 = frag_ld(&As[0][0], 40, wm * 32, lane);
    v16bf af1 = frag_ld(&As[0][0], 40, wm * 32 + 16, lane);
    v16bf bf0 = frag_ld(&BsT[0][0], 40, wn * 32, lane);
    v16bf bf1 = frag_ld(&BsT[0][0], 40, wn * 32 + 16, lane);
    acc[0][0] = wmma_bf16(af0, bf0, acc[0][0]);
    acc[0][1] = wmma_bf16(af0, bf1, acc[0][1]);
    acc[1][0] = wmma_bf16(af1, bf0, acc[1][0]);
    acc[1][1] = wmma_bf16(af1, bf1, acc[1][1]);
    __syncthreads();
  }
  int nl = lane & 15, hsel = lane >> 4;
#pragma unroll
  for (int i = 0; i < 2; ++i)
#pragma unroll
    for (int j = 0; j < 2; ++j)
#pragma unroll
      for (int v = 0; v < 8; ++v) {
        int m = wm * 32 + 16 * i + v + 8 * hsel;
        int n = nt * 64 + wn * 32 + 16 * j + nl;
        scores[((size_t)bhc * 128 + m) * KW + n] = acc[i][j][v];
      }
}

// ---------------------------------------------------------------- softmax(385)
__global__ __launch_bounds__(128) void softmax_kernel(
    const float* __restrict__ scores, const float* __restrict__ qf,
    const float* __restrict__ kf, const int* __restrict__ mask,
    bf16* __restrict__ probs, float* __restrict__ pg, int B_, int S_, int D_,
    int Hn) {
  const int Wn = 128, KW = 384;
  int nc = S_ / Wn;
  int idx = blockIdx.x;  // ((b*H+h)*nc + c)*128 + qi
  int qi = idx & (Wn - 1);
  int c = (idx / Wn) % nc;
  int hh = (idx / (Wn * nc)) % Hn;
  int b = idx / (Wn * nc * Hn);
  int tid = threadIdx.x;
  int s = c * Wn + qi;
  __shared__ float sv[KW + 1];
  __shared__ float red[128];

  float part = 0.f;
  if (tid < 64)
    part = qf[(size_t)(b * S_ + s) * D_ + hh * 64 + tid] *
           kf[(size_t)(b * S_) * D_ + hh * 64 + tid];
  red[tid] = part;
  __syncthreads();
  if (tid == 0) {
    float g = 0.f;
    for (int i = 0; i < 64; ++i) g += red[i];
    sv[KW] = (mask[b * S_] != 0) ? g : NEGV;
  }
  __syncthreads();

  const float* srow = scores + (size_t)idx * KW;
  for (int j = tid; j < KW; j += 128) {
    float v = srow[j];
    int gk = (c - 1) * Wn + j;
    bool rel = (j >= qi) && (j <= qi + 2 * Wn);
    bool kok = (gk >= 0) && (gk < S_) &&
               (mask[b * S_ + (gk < 0 ? 0 : (gk >= S_ ? S_ - 1 : gk))] != 0);
    sv[j] = (rel && kok) ? v : NEGV;
  }
  __syncthreads();
  float mx = -3.0e38f;
  for (int j = tid; j <= KW; j += 128) mx = fmaxf(mx, sv[j]);
  red[tid] = mx;
  __syncthreads();
  if (tid == 0) { float m2 = red[0]; for (int i = 1; i < 128; ++i) m2 = fmaxf(m2, red[i]); red[0] = m2; }
  __syncthreads();
  mx = red[0];
  __syncthreads();
  float psum = 0.f;
  for (int j = tid; j <= KW; j += 128) {
    float e = __expf(sv[j] - mx);
    sv[j] = e;
    psum += e;
  }
  red[tid] = psum;
  __syncthreads();
  if (tid == 0) { float t = 0; for (int i = 0; i < 128; ++i) t += red[i]; red[0] = t; }
  __syncthreads();
  float inv = 1.f / red[0];
  bf16* prow = probs + (size_t)idx * KW;
  for (int j = tid; j < KW; j += 128) prow[j] = (bf16)(sv[j] * inv);
  if (tid == 0) pg[(size_t)(b * Hn + hh) * S_ + s] = sv[KW] * inv;
}

// ---------------------------------------------------------------- P @ Vband
// per (b,h,chunk): out(128 x 64) = P(128 x 384) @ Vband(384 x 64) + pg*v0
__global__ __launch_bounds__(256) void pv_kernel(
    const bf16* __restrict__ probs, const bf16* __restrict__ vb,
    const float* __restrict__ pg, const float* __restrict__ vf,
    bf16* __restrict__ outb, int B_, int S_, int D_, int Hn) {
  const int Wn = 128, KW = 384;
  int nc = S_ / Wn;
  int bhc = blockIdx.x;
  int c = bhc % nc, bh = bhc / nc, hh = bh % Hn, b = bh / Hn;
  int tid = threadIdx.x, lane = tid & 31, w = tid >> 5;
  int wm = w & 3, wn = w >> 2;
  __shared__ alignas(16) bf16 As[128][40];
  __shared__ alignas(16) bf16 BsT[64][40];
  v8f acc[2][2] = {};
  for (int k0 = 0; k0 < KW; k0 += 32) {
    // A: probability rows, contiguous
#pragma unroll
    for (int it = 0; it < 2; ++it) {
      int idx = tid + it * 256;
      int r = idx >> 2, q = idx & 3;
      async_cp16(probs + ((size_t)bhc * 128 + r) * KW + k0 + q * 8,
                 &As[r][q * 8]);
    }
    // B^T: BsT[dim][key] = V[b, gk(key), hh, dim]; vector along dim, scatter
    {
      int kR = tid >> 3, q = tid & 7;  // 32 keys x 8 chunks of 8 dims
      int gk = (c - 1) * Wn + k0 + kR;
      int gkc = gk < 0 ? 0 : (gk >= S_ ? S_ - 1 : gk);
      union {
        uint4 u;
        bf16 h[8];
      } t;
      t.u = *(const uint4*)(vb + (size_t)(b * S_ + gkc) * D_ + hh * 64 + q * 8);
      bool ok = (gk >= 0) && (gk < S_);
#pragma unroll
      for (int e = 0; e < 8; ++e)
        BsT[q * 8 + e][kR] = ok ? t.h[e] : (bf16)0.0f;
    }
    async_wait0();
    __syncthreads();
    v16bf af0 = frag_ld(&As[0][0], 40, wm * 32, lane);
    v16bf af1 = frag_ld(&As[0][0], 40, wm * 32 + 16, lane);
    v16bf bf0 = frag_ld(&BsT[0][0], 40, wn * 32, lane);
    v16bf bf1 = frag_ld(&BsT[0][0], 40, wn * 32 + 16, lane);
    acc[0][0] = wmma_bf16(af0, bf0, acc[0][0]);
    acc[0][1] = wmma_bf16(af0, bf1, acc[0][1]);
    acc[1][0] = wmma_bf16(af1, bf0, acc[1][0]);
    acc[1][1] = wmma_bf16(af1, bf1, acc[1][1]);
    __syncthreads();
  }
  int nl = lane & 15, hsel = lane >> 4;
#pragma unroll
  for (int i = 0; i < 2; ++i)
#pragma unroll
    for (int j = 0; j < 2; ++j)
#pragma unroll
      for (int v = 0; v < 8; ++v) {
        int m = wm * 32 + 16 * i + v + 8 * hsel;
        int n = wn * 32 + 16 * j + nl;  // < 64
        int s = c * Wn + m;
        float val = acc[i][j][v] +
                    pg[(size_t)(b * Hn + hh) * S_ + s] *
                        vf[(size_t)(b * S_) * D_ + hh * 64 + n];
        outb[(size_t)(b * S_ + s) * D_ + hh * 64 + n] = (bf16)val;
      }
}

// ---------------------------------------------------------------- global attn (token 0)
__global__ __launch_bounds__(256) void global_attn_kernel(
    const float* __restrict__ qgf, const float* __restrict__ kgf,
    const float* __restrict__ vgf, const int* __restrict__ mask,
    bf16* __restrict__ outb, int B_, int S_, int D_, int Hn) {
  int hh = blockIdx.x % Hn, b = blockIdx.x / Hn;
  int tid = threadIdx.x;
  __shared__ float qsh[64];
  __shared__ float sc[2048];
  __shared__ float red[256];
  if (tid < 64) qsh[tid] = qgf[(size_t)b * D_ + hh * 64 + tid];
  __syncthreads();
  for (int s = tid; s < S_; s += 256) {
    float a = 0.f;
    const float* kr = kgf + (size_t)(b * S_ + s) * D_ + hh * 64;
#pragma unroll 8
    for (int d = 0; d < 64; ++d) a += qsh[d] * kr[d];
    sc[s] = (mask[b * S_ + s] != 0) ? a : NEGV;
  }
  __syncthreads();
  float mx = -3.0e38f;
  for (int s = tid; s < S_; s += 256) mx = fmaxf(mx, sc[s]);
  red[tid] = mx;
  __syncthreads();
  if (tid == 0) { float m = red[0]; for (int i = 1; i < 256; ++i) m = fmaxf(m, red[i]); red[0] = m; }
  __syncthreads();
  mx = red[0];
  __syncthreads();
  float ps = 0.f;
  for (int s = tid; s < S_; s += 256) {
    float e = __expf(sc[s] - mx);
    sc[s] = e;
    ps += e;
  }
  red[tid] = ps;
  __syncthreads();
  if (tid == 0) { float t = 0; for (int i = 0; i < 256; ++i) t += red[i]; red[0] = t; }
  __syncthreads();
  float inv = 1.f / red[0];
  if (tid < 64) {
    float o = 0.f;
    for (int s = 0; s < S_; ++s)
      o += sc[s] * vgf[(size_t)(b * S_ + s) * D_ + hh * 64 + tid];
    outb[(size_t)(b * S_) * D_ + hh * 64 + tid] = (bf16)(o * inv);
  }
}

// ---------------------------------------------------------------- classifier
__global__ void cls_kernel(const float* __restrict__ hf,
                           const float* __restrict__ Wc,
                           const float* __restrict__ bc,
                           const int* __restrict__ mask,
                           float* __restrict__ out, int B_, int S_, int D_) {
  int t = threadIdx.x;
  if (t < B_ * 3) {
    int b = t / 3, c = t % 3;
    float a = bc[c];
    for (int d = 0; d < D_; ++d) a += hf[(size_t)(b * S_) * D_ + d] * Wc[d * 3 + c];
    out[c * B_ + b] = 1.f / (1.f + __expf(-a));
  }
  if (t < B_) out[3 * B_ + t] = (float)mask[t * S_];
}

// ================================================================ host side
extern "C" void kernel_launch(void* const* d_in, const int* in_sizes, int n_in,
                              void* d_out, int out_size, void* d_ws,
                              size_t ws_size, hipStream_t stream) {
  (void)n_in; (void)out_size; (void)ws_size;
  const int* src = (const int*)d_in[0];
  const int* mask = (const int*)d_in[1];
  const float* emb_tok = (const float*)d_in[3];
  const float* emb_pos = (const float*)d_in[4];
  const float* eln_s = (const float*)d_in[5];
  const float* eln_b = (const float*)d_in[6];
  const float* Wq = (const float*)d_in[7];   const float* bq = (const float*)d_in[8];
  const float* Wk = (const float*)d_in[9];   const float* bk = (const float*)d_in[10];
  const float* Wv = (const float*)d_in[11];  const float* bv = (const float*)d_in[12];
  const float* Wo = (const float*)d_in[13];  const float* bo = (const float*)d_in[14];
  const float* Wqg = (const float*)d_in[15]; const float* bqg = (const float*)d_in[16];
  const float* Wkg = (const float*)d_in[17]; const float* bkg = (const float*)d_in[18];
  const float* Wvg = (const float*)d_in[19]; const float* bvg = (const float*)d_in[20];
  const float* ln1s = (const float*)d_in[21]; const float* ln1b = (const float*)d_in[22];
  const float* W1 = (const float*)d_in[23];  const float* b1 = (const float*)d_in[24];
  const float* W2 = (const float*)d_in[25];  const float* b2 = (const float*)d_in[26];
  const float* ln2s = (const float*)d_in[27]; const float* ln2b = (const float*)d_in[28];
  const float* Wcls = (const float*)d_in[29]; const float* bcls = (const float*)d_in[30];

  const int D = in_sizes[5];
  const int S = in_sizes[4] / D;
  const int B = in_sizes[0] / S;
  const int L = in_sizes[8] / D;
  const int FF = in_sizes[24] / L;
  const int H = 12, W = 128;
  const int nc = S / W;
  const int MS = B * S;
  const float alphaQ = 1.0f / sqrtf((float)(D / H));

  char* base = (char*)d_ws;
  size_t off = 0;
  auto alloc = [&](size_t bytes) -> void* {
    void* p = base + off;
    off = (off + bytes + 255) & ~(size_t)255;
    return p;
  };
  const size_t nDD = (size_t)L * D * D;
  const size_t nDF = (size_t)L * D * FF;
  bf16* wqB = (bf16*)alloc(nDD * 2);
  bf16* wkB = (bf16*)alloc(nDD * 2);
  bf16* wvB = (bf16*)alloc(nDD * 2);
  bf16* woB = (bf16*)alloc(nDD * 2);
  bf16* wqgB = (bf16*)alloc(nDD * 2);
  bf16* wkgB = (bf16*)alloc(nDD * 2);
  bf16* wvgB = (bf16*)alloc(nDD * 2);
  bf16* w1B = (bf16*)alloc(nDF * 2);
  bf16* w2B = (bf16*)alloc(nDF * 2);
  float* hf = (float*)alloc((size_t)MS * D * 4);
  bf16*  hb = (bf16*)alloc((size_t)MS * D * 2);
  float* qf = (float*)alloc((size_t)MS * D * 4);
  bf16*  qb = (bf16*)alloc((size_t)MS * D * 2);
  float* kf = (float*)alloc((size_t)MS * D * 4);
  bf16*  kb = (bf16*)alloc((size_t)MS * D * 2);
  float* vf = (float*)alloc((size_t)MS * D * 4);
  bf16*  vb = (bf16*)alloc((size_t)MS * D * 2);
  float* kgf = (float*)alloc((size_t)MS * D * 4);
  float* vgf = (float*)alloc((size_t)MS * D * 4);
  float* qgf = (float*)alloc((size_t)B * D * 4);
  float* scores = (float*)alloc((size_t)B * H * nc * 128 * 384 * 4);
  bf16*  probs = (bf16*)alloc((size_t)B * H * nc * 128 * 384 * 2);
  float* pgf = (float*)alloc((size_t)B * H * S * 4);
  bf16*  attb = (bf16*)alloc((size_t)MS * D * 2);
  float* tf = (float*)alloc((size_t)MS * D * 4);
  bf16*  f1b = (bf16*)alloc((size_t)MS * FF * 2);

  f32_to_bf16_kernel<<<1024, 256, 0, stream>>>(Wq, wqB, nDD);
  f32_to_bf16_kernel<<<1024, 256, 0, stream>>>(Wk, wkB, nDD);
  f32_to_bf16_kernel<<<1024, 256, 0, stream>>>(Wv, wvB, nDD);
  f32_to_bf16_kernel<<<1024, 256, 0, stream>>>(Wo, woB, nDD);
  f32_to_bf16_kernel<<<1024, 256, 0, stream>>>(Wqg, wqgB, nDD);
  f32_to_bf16_kernel<<<1024, 256, 0, stream>>>(Wkg, wkgB, nDD);
  f32_to_bf16_kernel<<<1024, 256, 0, stream>>>(Wvg, wvgB, nDD);
  f32_to_bf16_kernel<<<1024, 256, 0, stream>>>(W1, w1B, nDF);
  f32_to_bf16_kernel<<<1024, 256, 0, stream>>>(W2, w2B, nDF);

  embed_ln_kernel<<<MS, 256, 0, stream>>>(src, emb_tok, emb_pos, eln_s, eln_b,
                                          hf, hb, S, D);

  dim3 gD((D + 63) / 64, (MS + 127) / 128);
  dim3 gF((FF + 63) / 64, (MS + 127) / 128);
  for (int l = 0; l < L; ++l) {
    size_t wOff = (size_t)l * D * D;
    size_t fOff = (size_t)l * D * FF;
    gemm_bf16_kernel<<<gD, 256, 0, stream>>>(hb, D, wqB + wOff, D, bq + l * D,
                                             qf, qb, D, MS, D, D, alphaQ, 0);
    gemm_bf16_kernel<<<gD, 256, 0, stream>>>(hb, D, wkB + wOff, D, bk + l * D,
                                             kf, kb, D, MS, D, D, 1.0f, 0);
    gemm_bf16_kernel<<<gD, 256, 0, stream>>>(hb, D, wvB + wOff, D, bv + l * D,
                                             vf, vb, D, MS, D, D, 1.0f, 0);
    gemm_bf16_kernel<<<gD, 256, 0, stream>>>(hb, D, wkgB + wOff, D, bkg + l * D,
                                             kgf, (bf16*)nullptr, D, MS, D, D, 1.0f, 0);
    gemm_bf16_kernel<<<gD, 256, 0, stream>>>(hb, D, wvgB + wOff, D, bvg + l * D,
                                             vgf, (bf16*)nullptr, D, MS, D, D, 1.0f, 0);
    gemm_bf16_kernel<<<dim3((D + 63) / 64, 1), 256, 0, stream>>>(
        hb, S * D, wqgB + wOff, D, bqg + l * D, qgf, (bf16*)nullptr, D, B, D, D,
        alphaQ, 0);
    qk_scores_kernel<<<dim3(6, B * H * nc), 256, 0, stream>>>(qb, kb, scores,
                                                              B, S, D, H);
    softmax_kernel<<<B * H * S, 128, 0, stream>>>(scores, qf, kf, mask, probs,
                                                  pgf, B, S, D, H);
    pv_kernel<<<B * H * nc, 256, 0, stream>>>(probs, vb, pgf, vf, attb, B, S,
                                              D, H);
    global_attn_kernel<<<B * H, 256, 0, stream>>>(qgf, kgf, vgf, mask, attb, B,
                                                  S, D, H);
    gemm_bf16_kernel<<<gD, 256, 0, stream>>>(attb, D, woB + wOff, D, bo + l * D,
                                             tf, (bf16*)nullptr, D, MS, D, D, 1.0f, 0);
    add_ln_kernel<<<MS, 256, 0, stream>>>(hf, tf, ln1s + l * D, ln1b + l * D,
                                          hf, hb, D);
    gemm_bf16_kernel<<<gF, 256, 0, stream>>>(hb, D, w1B + fOff, FF,
                                             b1 + (size_t)l * FF,
                                             (float*)nullptr, f1b, FF, MS, FF, D,
                                             1.0f, 1);
    gemm_bf16_kernel<<<gD, 256, 0, stream>>>(f1b, FF, w2B + fOff, D, b2 + l * D,
                                             tf, (bf16*)nullptr, D, MS, D, FF,
                                             1.0f, 0);
    add_ln_kernel<<<MS, 256, 0, stream>>>(hf, tf, ln2s + l * D, ln2b + l * D,
                                          hf, hb, D);
  }
  cls_kernel<<<1, 32, 0, stream>>>(hf, Wcls, bcls, mask, (float*)d_out, B, S, D);
}
